// Spatial_Attention_layer_73572789780804
// MI455X (gfx1250) — matrix operand order repaired and checked by reading, not compiled, not run on hardware
//
#include <hip/hip_runtime.h>
#include <hip/hip_bf16.h>
#include <math.h>

typedef __bf16 bf16_t;
typedef __attribute__((ext_vector_type(16))) __bf16 v16bf;
typedef __attribute__((ext_vector_type(8)))  __bf16 v8bf;
typedef __attribute__((ext_vector_type(8)))  float  v8f;
typedef int v4i_vs __attribute__((vector_size(16)));   // matches builtin param type

#define GLOBAL_AS __attribute__((address_space(1)))
#define LDS_AS    __attribute__((address_space(3)))

#define NB 32      // batch
#define NN 2048    // N
#define NF 64      // F
#define NT 24      // T
#define KP 32      // padded K for T (zeros in 24..31)

#define APITCH 72  // LDS row pitch (elements) for A tile (144B, 16B aligned, bank spread)
#define BPITCH 72

#if defined(__gfx1250__) && __has_builtin(__builtin_amdgcn_global_load_async_to_lds_b128)
#define HAVE_ASYNC_LDS 1
#else
#define HAVE_ASYNC_LDS 0
#endif

__device__ __forceinline__ void async_wait_all() {
#if defined(__HIP_DEVICE_COMPILE__)
  asm volatile("s_wait_asynccnt 0x0" ::: "memory");
#endif
}

// 16-byte global -> LDS copy; async (ASYNCcnt-tracked) when available.
__device__ __forceinline__ void copy16_g2l(const bf16_t* g, bf16_t* l) {
#if HAVE_ASYNC_LDS
  __builtin_amdgcn_global_load_async_to_lds_b128(
      (GLOBAL_AS v4i_vs*)(v4i_vs*)g,
      (LDS_AS v4i_vs*)(v4i_vs*)l, 0, 0);
#else
  *(uint4*)l = *(const uint4*)g;
#endif
}

__device__ __forceinline__ unsigned short f32_to_bf16_bits(float f) {
  union { float f; unsigned u; } v; v.f = f;
  unsigned u = v.u;
  unsigned r = (u + 0x7fffu + ((u >> 16) & 1u)) >> 16;  // RNE
  return (unsigned short)r;
}

// 16-bit A-matrix 16x32 fragment: lanes 0-15 hold K=0..7 (v0-3) and 16..23 (v4-7),
// lanes 16-31 hold K=8..15 and 24..31.  Caller passes p already offset by +8*half.
__device__ __forceinline__ v16bf load_fragA(const bf16_t* p) {
  v8bf lo = *(const v8bf*)(p);        // K base+0..7
  v8bf hi = *(const v8bf*)(p + 16);   // K base+16..23
  v16bf r;
#pragma unroll
  for (int i = 0; i < 8; ++i) { r[i] = lo[i]; r[i + 8] = hi[i]; }
  return r;
}

// B-matrix 32x16 fragment from transposed storage [col][k] (k contiguous):
// lanes 0-15: col=lane, K=0..7,16..23 ; lanes 16-31: col=lane-16, K=8..15,24..31.
// Caller passes p = row(col) base + 8*half.
__device__ __forceinline__ v16bf load_fragB(const bf16_t* p) {
  v8bf lo = *(const v8bf*)(p);
  v8bf hi = *(const v8bf*)(p + 16);
  v16bf r;
#pragma unroll
  for (int i = 0; i < 8; ++i) { r[i] = lo[i]; r[i + 8] = hi[i]; }
  return r;
}

// ---------------- Kernel 1: Vs fp32 -> bf16 ----------------
__global__ __launch_bounds__(256) void k_cvt_vs(const float* __restrict__ vs,
                                                unsigned short* __restrict__ vsb) {
  size_t i = (size_t)blockIdx.x * blockDim.x + threadIdx.x;
  if (i < (size_t)NN * NN) vsb[i] = f32_to_bf16_bits(vs[i]);
}

// ---------------- Kernel 2: stage A ----------------
// Per (b,n): lhs[b,n,t] = sum_f (sum_t' x[b,n,f,t']*W1[t']) * W2[f,t]
//            rhs[b,t,n] = sum_f W3[f]*x[b,n,f,t]
// Written bf16, K padded to 32 with zeros, rhs stored transposed as [b][n][t].
__global__ __launch_bounds__(64) void k_stageA(const float* __restrict__ x,
                                               const float* __restrict__ W1,
                                               const float* __restrict__ W2,
                                               const float* __restrict__ W3,
                                               unsigned short* __restrict__ lhsA,
                                               unsigned short* __restrict__ rhsT) {
  __shared__ float s_l[64][NT];
  __shared__ float s_r[64][NT];
  int bn = blockIdx.x;            // b*NN + n
  int f  = threadIdx.x;           // 0..63
  const float* xr = x + ((size_t)bn * NF + f) * NT;
  float xrow[NT];
#pragma unroll
  for (int t = 0; t < NT; ++t) xrow[t] = xr[t];
  float a = 0.f;
#pragma unroll
  for (int t = 0; t < NT; ++t) a += xrow[t] * W1[t];
  float w3 = W3[f];
  const float* w2r = W2 + f * NT;
#pragma unroll
  for (int t = 0; t < NT; ++t) {
    s_l[f][t] = a * w2r[t];
    s_r[f][t] = w3 * xrow[t];
  }
  __syncthreads();
  for (int s = 32; s > 0; s >>= 1) {
    if (f < s) {
#pragma unroll
      for (int t = 0; t < NT; ++t) {
        s_l[f][t] += s_l[f + s][t];
        s_r[f][t] += s_r[f + s][t];
      }
    }
    __syncthreads();
  }
  unsigned short* lo = lhsA + (size_t)bn * KP;
  unsigned short* ro = rhsT + (size_t)bn * KP;
  if (f < KP) {
    lo[f] = (f < NT) ? f32_to_bf16_bits(s_l[0][f]) : (unsigned short)0;
  } else {
    int t = f - KP;
    ro[t] = (t < NT) ? f32_to_bf16_bits(s_r[0][t]) : (unsigned short)0;
  }
}

// ---------------- Kernel 3: stage B ----------------
// product[b,n,m] = sum_t lhs[b,n,t]*rhs[b,t,m]  (single K=32 WMMA step)
// sig = sigmoid(product + bs[n,m]) -> bf16, stored transposed sigT[b][m][n].
__global__ __launch_bounds__(256) void k_stageB(const bf16_t* __restrict__ lhsA,
                                                const bf16_t* __restrict__ rhsT,
                                                const float* __restrict__ bs,
                                                unsigned short* __restrict__ sigT) {
  int wave = threadIdx.x >> 5;    // wave32
  int lane = threadIdx.x & 31;
  int r    = lane & 15;
  int half = lane >> 4;
  int gid  = blockIdx.x;          // NB * 2048 blocks
  int b    = gid >> 11;
  int tile = gid & 2047;
  int frag = tile * 8 + wave;     // 0..16383 per batch
  int n0   = (frag >> 7) << 4;
  int m0   = (frag & 127) << 4;

  const bf16_t* Ab = lhsA + (size_t)b * NN * KP;
  const bf16_t* Bb = rhsT + (size_t)b * NN * KP;
  v16bf a  = load_fragA(Ab + (size_t)(n0 + r) * KP + 8 * half);
  v16bf bf = load_fragB(Bb + (size_t)(m0 + r) * KP + 8 * half);
  v8f c = {};
  c = __builtin_amdgcn_wmma_f32_16x16x32_bf16(false, a, false, bf, (short)0, c,
                                              false, false);
  int mcol  = m0 + r;
  int nbase = n0 + 8 * half;
  unsigned short h[8];
#pragma unroll
  for (int q = 0; q < 8; ++q) {
    float v = c[q] + bs[(size_t)(nbase + q) * NN + mcol];
    float e = __expf(-v);
    float s = __builtin_amdgcn_rcpf(1.0f + e);   // fast v_rcp_f32, enough for bf16
    h[q] = f32_to_bf16_bits(s);
  }
  uint4 pk;
  pk.x = (unsigned)h[0] | ((unsigned)h[1] << 16);
  pk.y = (unsigned)h[2] | ((unsigned)h[3] << 16);
  pk.z = (unsigned)h[4] | ((unsigned)h[5] << 16);
  pk.w = (unsigned)h[6] | ((unsigned)h[7] << 16);
  *(uint4*)(sigT + (size_t)b * NN * NN + (size_t)mcol * NN + nbase) = pk;
}

// ---------------- Kernel 4: stage C ----------------
// S[b] = Vs_bf16 (2048x2048) @ sig[b] (2048x2048) -> fp32 into d_out.
// Block = 8 waves -> 128x64 tile; wave computes 32x32 (2x2 fragments).
// Tiles double-buffered in LDS, filled with async global->LDS copies.
__device__ __forceinline__ void store_frag(float* ob, int n0, int m0, int r,
                                           int half, const v8f& c) {
  float* p = ob + (size_t)(n0 + 8 * half) * NN + m0 + r;
#pragma unroll
  for (int q = 0; q < 8; ++q) p[(size_t)q * NN] = c[q];
}

__global__ __launch_bounds__(256) void k_stageC(const bf16_t* __restrict__ VsB,
                                                const bf16_t* __restrict__ sigT,
                                                float* __restrict__ out) {
  __shared__ __align__(16) bf16_t sA[2][128][APITCH];  // 128 rows(n) x 64 k
  __shared__ __align__(16) bf16_t sB[2][64][BPITCH];   // 64 rows(m)  x 64 k

  int tid  = threadIdx.x;
  int wave = tid >> 5;
  int lane = tid & 31;
  int r    = lane & 15;
  int half = lane >> 4;
  int b    = blockIdx.z;
  int wn   = wave >> 1, wm = wave & 1;
  int n0b  = blockIdx.y * 128;
  int m0b  = blockIdx.x * 64;

  const bf16_t* Ag = VsB + (size_t)n0b * NN;                         // A rows n0b..+127
  const bf16_t* Bg = sigT + (size_t)b * NN * NN + (size_t)m0b * NN;  // B rows m0b..+63

  // per-thread transfer coordinates (16B granules)
  int arow0 = tid >> 3, aseg = (tid & 7) * 8;   // + j*32 rows, j=0..3
  int brow0 = tid >> 3, bseg = (tid & 7) * 8;   // + j*32 rows, j=0..1

  // preload buffer 0
#pragma unroll
  for (int j = 0; j < 4; ++j)
    copy16_g2l(Ag + (size_t)(arow0 + 32 * j) * NN + aseg, &sA[0][arow0 + 32 * j][aseg]);
#pragma unroll
  for (int j = 0; j < 2; ++j)
    copy16_g2l(Bg + (size_t)(brow0 + 32 * j) * NN + bseg, &sB[0][brow0 + 32 * j][bseg]);
  async_wait_all();
  __syncthreads();

  v8f c00 = {}, c01 = {}, c10 = {}, c11 = {};
  for (int kk = 0; kk < NN; kk += 64) {
    int cur = (kk >> 6) & 1;
    if (kk + 64 < NN) {
      int nxt = cur ^ 1;
      int kn  = kk + 64;
#pragma unroll
      for (int j = 0; j < 4; ++j)
        copy16_g2l(Ag + (size_t)(arow0 + 32 * j) * NN + kn + aseg,
                   &sA[nxt][arow0 + 32 * j][aseg]);
#pragma unroll
      for (int j = 0; j < 2; ++j)
        copy16_g2l(Bg + (size_t)(brow0 + 32 * j) * NN + kn + bseg,
                   &sB[nxt][brow0 + 32 * j][bseg]);
    }
#pragma unroll
    for (int ks = 0; ks < 64; ks += 32) {
      v16bf a0 = load_fragA(&sA[cur][wn * 32 + r][ks + 8 * half]);
      v16bf a1 = load_fragA(&sA[cur][wn * 32 + 16 + r][ks + 8 * half]);
      v16bf b0 = load_fragB(&sB[cur][wm * 32 + r][ks + 8 * half]);
      v16bf b1 = load_fragB(&sB[cur][wm * 32 + 16 + r][ks + 8 * half]);
      c00 = __builtin_amdgcn_wmma_f32_16x16x32_bf16(false, a0, false, b0, (short)0, c00, false, false);
      c01 = __builtin_amdgcn_wmma_f32_16x16x32_bf16(false, a0, false, b1, (short)0, c01, false, false);
      c10 = __builtin_amdgcn_wmma_f32_16x16x32_bf16(false, a1, false, b0, (short)0, c10, false, false);
      c11 = __builtin_amdgcn_wmma_f32_16x16x32_bf16(false, a1, false, b1, (short)0, c11, false, false);
    }
    async_wait_all();   // next buffer resident before anyone proceeds
    __syncthreads();    // everyone done reading cur before it is refilled
  }

  float* ob = out + (size_t)b * NN * NN;
  int n0 = n0b + wn * 32;
  int m0 = m0b + wm * 32;
  store_frag(ob, n0,      m0,      r, half, c00);
  store_frag(ob, n0,      m0 + 16, r, half, c01);
  store_frag(ob, n0 + 16, m0,      r, half, c10);
  store_frag(ob, n0 + 16, m0 + 16, r, half, c11);
}

// ---------------- Kernel 5/6: column softmax over axis n ----------------
__global__ __launch_bounds__(256) void k_sm1(const float* __restrict__ S,
                                             float* __restrict__ cmax,
                                             float* __restrict__ csum) {
  int bidx = blockIdx.x;                 // NB * 8 blocks
  int b = bidx >> 3;
  int m = ((bidx & 7) << 8) + threadIdx.x;
  const float* col = S + (size_t)b * NN * NN + m;
  float M = -3.4e38f, Z = 0.f;
  for (int n = 0; n < NN; ++n) {
    float v  = col[(size_t)n * NN];
    float nm = fmaxf(M, v);
    Z = Z * __expf(M - nm) + __expf(v - nm);
    M = nm;
  }
  cmax[b * NN + m] = M;
  csum[b * NN + m] = Z;
}

__global__ __launch_bounds__(256) void k_sm2(float* __restrict__ S,
                                             const float* __restrict__ cmax,
                                             const float* __restrict__ csum) {
  int bidx = blockIdx.x;
  int b = bidx >> 3;
  int m = ((bidx & 7) << 8) + threadIdx.x;
  float* col = S + (size_t)b * NN * NN + m;
  float M   = cmax[b * NN + m];
  float inv = 1.0f / csum[b * NN + m];
  for (int n = 0; n < NN; ++n) {
    size_t off = (size_t)n * NN;
    col[off] = __expf(col[off] - M) * inv;
  }
}

extern "C" void kernel_launch(void* const* d_in, const int* in_sizes, int n_in,
                              void* d_out, int out_size, void* d_ws, size_t ws_size,
                              hipStream_t stream) {
  const float* x  = (const float*)d_in[0];
  const float* W1 = (const float*)d_in[1];
  const float* W2 = (const float*)d_in[2];
  const float* W3 = (const float*)d_in[3];
  const float* bs = (const float*)d_in[4];
  const float* Vs = (const float*)d_in[5];
  float* out = (float*)d_out;
  char* ws = (char*)d_ws;

  size_t offVs  = 0;
  size_t offL   = offVs  + (size_t)NN * NN * 2;        // Vs bf16: 8 MB
  size_t offR   = offL   + (size_t)NB * NN * KP * 2;   // lhs bf16: 4 MB
  size_t offSig = offR   + (size_t)NB * NN * KP * 2;   // rhsT bf16: 4 MB
  size_t offMax = offSig + (size_t)NB * NN * NN * 2;   // sigT bf16: 268 MB
  size_t offSum = offMax + (size_t)NB * NN * 4;

  unsigned short* VsB_w  = (unsigned short*)(ws + offVs);
  unsigned short* lhs_w  = (unsigned short*)(ws + offL);
  unsigned short* rhs_w  = (unsigned short*)(ws + offR);
  unsigned short* sig_w  = (unsigned short*)(ws + offSig);
  float*          cmax   = (float*)(ws + offMax);
  float*          csum   = (float*)(ws + offSum);
  const bf16_t* VsB = (const bf16_t*)(ws + offVs);
  const bf16_t* lhs = (const bf16_t*)(ws + offL);
  const bf16_t* rhs = (const bf16_t*)(ws + offR);
  const bf16_t* sig = (const bf16_t*)(ws + offSig);

  k_cvt_vs<<<(NN * NN + 255) / 256, 256, 0, stream>>>(Vs, VsB_w);
  k_stageA<<<NB * NN, 64, 0, stream>>>(x, W1, W2, W3, lhs_w, rhs_w);
  k_stageB<<<NB * 2048, 256, 0, stream>>>(lhs, rhs, bs, sig_w);
  dim3 gC(NN / 64, NN / 128, NB);
  k_stageC<<<gC, 256, 0, stream>>>(VsB, sig, out);
  k_sm1<<<NB * 8, 256, 0, stream>>>(out, cmax, csum);
  k_sm2<<<NB * 8, 256, 0, stream>>>(out, cmax, csum);
}
